// MultiScaleGRU_48541720379654
// MI455X (gfx1250) — matrix operand order repaired and checked by reading, not compiled, not run on hardware
//
#include <hip/hip_runtime.h>
#include <hip/hip_bf16.h>

// MultiScaleGRU fused kernel for MI455X (gfx1250, wave32, WMMA).
//
// Shapes: B=256, S=64, F=4096, H=256, L=4, O=16.
// Phase 1 GEMM: X[16384,256] = AW[16384,4096] x CC[4096,256]  (34.4 GFLOP, 268MB read)
// Phase 2 GEMM: gi[16384,768] = X x w_ih^T                     (6.4 GFLOP, X kept in LDS)
//
// Batch-invariant operands are preprocessed once into workspace:
//   ghb[s][j]   = CC[s/4]·w_hh[j] + b_hh[j] (+ b_ih[j] for r,z)
//   CCswz       = current_centers as f16 in WMMA B-fragment layout
//   Wswz        = w_ih^T as f16 in WMMA B-fragment layout

typedef __attribute__((ext_vector_type(16))) _Float16 v16h;
typedef __attribute__((ext_vector_type(8)))  _Float16 v8h;
typedef __attribute__((ext_vector_type(8)))  float    v8f;

#define HID 256
#define FEA 4096
#define NBATCH 256
#define NSLOT 64
#define G3 768

__device__ __forceinline__ float fsigmoid(float x) {
    return 1.0f / (1.0f + __expf(-x));
}
__device__ __forceinline__ float ftanh(float x) {
    x = fminf(fmaxf(x, -15.0f), 15.0f);
    float e = __expf(2.0f * x);
    return (e - 1.0f) / (e + 1.0f);
}

// Pack 4x float4 (two 8-float K-groups) into an f16 A-fragment register.
__device__ __forceinline__ v16h pack16(float4 a0, float4 a1, float4 a2, float4 a3) {
    v16h r;
    r[0]  = (_Float16)a0.x; r[1]  = (_Float16)a0.y; r[2]  = (_Float16)a0.z; r[3]  = (_Float16)a0.w;
    r[4]  = (_Float16)a1.x; r[5]  = (_Float16)a1.y; r[6]  = (_Float16)a1.z; r[7]  = (_Float16)a1.w;
    r[8]  = (_Float16)a2.x; r[9]  = (_Float16)a2.y; r[10] = (_Float16)a2.z; r[11] = (_Float16)a2.w;
    r[12] = (_Float16)a3.x; r[13] = (_Float16)a3.y; r[14] = (_Float16)a3.z; r[15] = (_Float16)a3.w;
    return r;
}

// ---------------------------------------------------------------------------
// Kernel 0a: ghb[s][j] = sum_h CC[s/4][h]*w_hh[j][h] + b_hh[j] + (j<512 ? b_ih[j] : 0)
// ---------------------------------------------------------------------------
__global__ void msgru_ghb_kernel(const float* __restrict__ CC,
                                 const float* __restrict__ w_hh,
                                 const float* __restrict__ b_ih,
                                 const float* __restrict__ b_hh,
                                 float* __restrict__ ghb) {
    __shared__ float hrow[HID];
    const int s = blockIdx.x;
    const int o = s >> 2;
    if (threadIdx.x < HID) hrow[threadIdx.x] = CC[(size_t)o * HID + threadIdx.x];
    __syncthreads();

    const int j = threadIdx.x;            // 0..767
    const float4* wr = (const float4*)(w_hh + (size_t)j * HID);
    float acc = 0.0f;
#pragma unroll 4
    for (int h4 = 0; h4 < HID / 4; ++h4) {
        float4 w = wr[h4];
        const float* hp = &hrow[h4 * 4];
        acc += w.x * hp[0] + w.y * hp[1] + w.z * hp[2] + w.w * hp[3];
    }
    float v = acc + b_hh[j];
    if (j < 512) v += b_ih[j];            // fold b_ih into r,z pre-activations only
    ghb[(size_t)s * G3 + j] = v;
}

// ---------------------------------------------------------------------------
// Kernel 0b: CCswz[kb][n][e] = (f16)CC[kb*32 + e][n]   (B-fragment layout)
// Lane (col n, half h) of K-block kb later reads 16 contiguous halfs at
// ((kb*256 + n)*32 + 16h). 128 blocks x 256 threads; LDS transpose so both
// the global read and the global write are coalesced.
// ---------------------------------------------------------------------------
__global__ void msgru_swz_cc_kernel(const float* __restrict__ CC,
                                    _Float16* __restrict__ dst) {
    __shared__ float tile[32 * HID];
    const int kb = blockIdx.x;            // 0..127
    const int n  = threadIdx.x;           // 0..255
#pragma unroll 8
    for (int e = 0; e < 32; ++e)
        tile[e * HID + n] = CC[(size_t)(kb * 32 + e) * HID + n];
    __syncthreads();

    _Float16* dp = dst + ((size_t)kb * HID + n) * 32;
#pragma unroll
    for (int q = 0; q < 4; ++q) {
        v8h v;
#pragma unroll
        for (int e = 0; e < 8; ++e) v[e] = (_Float16)tile[(q * 8 + e) * HID + n];
        *(v8h*)(dp + q * 8) = v;
    }
}

// ---------------------------------------------------------------------------
// Kernel 0c: Wswz[kb][n][e] = (f16)w_ih[n*256 + kb*32 + e]  (B-fragment layout
// of w_ih^T). Source is already contiguous per lane, so no LDS needed.
// 24 blocks x 256 threads (8 K-blocks x 768 columns).
// ---------------------------------------------------------------------------
__global__ void msgru_swz_wih_kernel(const float* __restrict__ w_ih,
                                     _Float16* __restrict__ dst) {
    const int idx = blockIdx.x * 256 + threadIdx.x;   // 0..6143
    const int n   = idx % G3;
    const int kb  = idx / G3;                         // 0..7
    const float* sp = w_ih + (size_t)n * HID + kb * 32;
    _Float16*    dp = dst + ((size_t)kb * G3 + n) * 32;
#pragma unroll
    for (int q = 0; q < 4; ++q) {
        float4 f0 = *(const float4*)(sp + q * 8);
        float4 f1 = *(const float4*)(sp + q * 8 + 4);
        v8h v;
        v[0] = (_Float16)f0.x; v[1] = (_Float16)f0.y; v[2] = (_Float16)f0.z; v[3] = (_Float16)f0.w;
        v[4] = (_Float16)f1.x; v[5] = (_Float16)f1.y; v[6] = (_Float16)f1.z; v[7] = (_Float16)f1.w;
        *(v8h*)(dp + q * 8) = v;
    }
}

// ---------------------------------------------------------------------------
// Kernel 1: fused  X = AW x CC  ->  GRU gates  ->  level-mean  ->  out[O,B,H]
// Grid: 256 batches x 2 slot-blocks of 32 slots. Block: 512 threads = 16 waves.
// Wave w owns N-columns [16w,16w+16) and MT=2 row-tiles of 16 slots each.
// ---------------------------------------------------------------------------
#define MT 2
#define XSTRIDE 272   // padded LDS row stride (floats) to spread banks

__global__ void __launch_bounds__(512)
msgru_fused_kernel(const float* __restrict__ AW,      // [B,S,F]
                   const _Float16* __restrict__ CCswz,// [F/32,256,32] f16 B-frags
                   const _Float16* __restrict__ Wswz, // [8,768,32]    f16 B-frags
                   const float* __restrict__ CC,      // [F,H] (for h state)
                   const float* __restrict__ b_ih,    // [3H]
                   const float* __restrict__ ghb,     // [S,3H] from kernel 0a
                   float* __restrict__ out) {         // [O,B,H]
    __shared__ __align__(16) float Xlds[MT * 16 * XSTRIDE];

    const int b    = blockIdx.x >> 1;
    const int s0   = (blockIdx.x & 1) * (MT * 16);
    const int wave = threadIdx.x >> 5;     // 0..15
    const int lane = threadIdx.x & 31;
    const int l15  = lane & 15;
    const int half = lane >> 4;
    const int n0   = wave * 16;
    const int col  = n0 + l15;

    // ---------------- Phase 1: X tile = AW rows x CC, f16 WMMA, f32 acc ----
    v8f acc[MT];
#pragma unroll
    for (int mt = 0; mt < MT; ++mt) acc[mt] = (v8f){0.f,0.f,0.f,0.f,0.f,0.f,0.f,0.f};

    const float* Abase = AW + ((size_t)b * NSLOT + s0) * FEA;
    const _Float16* Bcol = CCswz + (size_t)col * 32 + 16 * half;   // + kb32*256*32

    for (int kb = 0; kb < FEA; kb += 32) {
        // B fragment: one contiguous 32B chunk per lane (pre-swizzled f16).
        v16h bfrag = *(const v16h*)(Bcol + (size_t)(kb >> 5) * (HID * 32));

#pragma unroll
        for (int mt = 0; mt < MT; ++mt) {
            // A fragment: lane row = mt*16+l15; K range {kb+16h..+15} contiguous.
            const float* ar = Abase + (size_t)(mt * 16 + l15) * FEA + kb + 16 * half;
            float4 a0 = *(const float4*)(ar);
            float4 a1 = *(const float4*)(ar + 4);
            float4 a2 = *(const float4*)(ar + 8);
            float4 a3 = *(const float4*)(ar + 12);
            v16h afrag = pack16(a0, a1, a2, a3);
            acc[mt] = __builtin_amdgcn_wmma_f32_16x16x32_f16(
                false, afrag, false, bfrag, (short)0, acc[mt], false, false);
        }
    }

    // Spill X tiles to LDS (row rowm = i + 8*half, col = n0 + l15).
#pragma unroll
    for (int mt = 0; mt < MT; ++mt)
#pragma unroll
        for (int i = 0; i < 8; ++i)
            Xlds[(mt * 16 + i + 8 * half) * XSTRIDE + col] = acc[mt][i];
    __syncthreads();

    // ---------------- Phase 2: gi = X x w_ih^T  (K = 256) ------------------
    // Gate g columns j = g*256 + col so r/z/n land in the same lanes/VGPRs.
    v8f gacc[MT][3];
#pragma unroll
    for (int mt = 0; mt < MT; ++mt)
#pragma unroll
        for (int g = 0; g < 3; ++g)
            gacc[mt][g] = (v8f){0.f,0.f,0.f,0.f,0.f,0.f,0.f,0.f};

    for (int kb = 0; kb < HID; kb += 32) {
        v16h bf[3];
#pragma unroll
        for (int g = 0; g < 3; ++g)
            bf[g] = *(const v16h*)(Wswz +
                    ((size_t)(kb >> 5) * G3 + g * HID + col) * 32 + 16 * half);
#pragma unroll
        for (int mt = 0; mt < MT; ++mt) {
            const float* xr = &Xlds[(mt * 16 + l15) * XSTRIDE + kb + 16 * half];
            float4 x0 = *(const float4*)(xr);
            float4 x1 = *(const float4*)(xr + 4);
            float4 x2 = *(const float4*)(xr + 8);
            float4 x3 = *(const float4*)(xr + 12);
            v16h af = pack16(x0, x1, x2, x3);
#pragma unroll
            for (int g = 0; g < 3; ++g)
                gacc[mt][g] = __builtin_amdgcn_wmma_f32_16x16x32_f16(
                    false, af, false, bf[g], (short)0, gacc[mt][g], false, false);
        }
    }

    // ---------------- Gates + level-mean + store ---------------------------
    const float bihn = b_ih[512 + col];

#pragma unroll
    for (int mt = 0; mt < MT; ++mt) {
        const int obase = ((s0 + mt * 16) >> 2) + half * 2;   // object for rows i<4
        float hval0 = CC[(size_t)obase * HID + col];          // h = CC row (per object)
        float hval1 = CC[(size_t)(obase + 1) * HID + col];
        float osum0 = 0.0f, osum1 = 0.0f;
#pragma unroll
        for (int i = 0; i < 8; ++i) {
            const int s = s0 + mt * 16 + i + 8 * half;
            const float* gb = ghb + (size_t)s * G3;
            float r  = fsigmoid(gacc[mt][0][i] + gb[col]);
            float z  = fsigmoid(gacc[mt][1][i] + gb[HID + col]);
            float nn = ftanh(gacc[mt][2][i] + bihn + r * gb[2 * HID + col]);
            float h  = (i < 4) ? hval0 : hval1;
            float hnew = (1.0f - z) * nn + z * h;
            if (i < 4) osum0 += hnew; else osum1 += hnew;
        }
        out[((size_t)obase * NBATCH + b) * HID + col]       = osum0 * 0.25f;
        out[((size_t)(obase + 1) * NBATCH + b) * HID + col] = osum1 * 0.25f;
    }
}

// ---------------------------------------------------------------------------
extern "C" void kernel_launch(void* const* d_in, const int* in_sizes, int n_in,
                              void* d_out, int out_size, void* d_ws, size_t ws_size,
                              hipStream_t stream) {
    (void)in_sizes; (void)n_in; (void)out_size; (void)ws_size;
    const float* AW   = (const float*)d_in[0];  // [256,64,4096]
    const float* CC   = (const float*)d_in[1];  // [4096,256]
    const float* Wih  = (const float*)d_in[2];  // [768,256]
    const float* Whh  = (const float*)d_in[3];  // [768,256]
    const float* bih  = (const float*)d_in[4];  // [768]
    const float* bhh  = (const float*)d_in[5];  // [768]
    float*       outp = (float*)d_out;          // [16,256,256]

    // Workspace layout (all 16B aligned):
    //   [0, 192KB)           ghb   : float[64*768]
    //   [192KB, 192KB+2MB)   CCswz : f16[4096*256]
    //   [+384KB)             Wswz  : f16[8*768*32]
    float*    ghb   = (float*)d_ws;
    _Float16* CCswz = (_Float16*)((char*)d_ws + 196608);
    _Float16* Wswz  = (_Float16*)((char*)d_ws + 196608 + 2097152);

    msgru_ghb_kernel    <<<NSLOT, G3,  0, stream>>>(CC, Whh, bih, bhh, ghb);
    msgru_swz_cc_kernel <<<FEA / 32, HID, 0, stream>>>(CC, CCswz);
    msgru_swz_wih_kernel<<<(8 * G3) / 256, 256, 0, stream>>>(Wih, Wswz);
    msgru_fused_kernel  <<<NBATCH * (NSLOT / (MT * 16)), 512, 0, stream>>>(
        AW, CCswz, Wswz, CC, bih, ghb, outp);
}
